// Net_17308718203169
// MI455X (gfx1250) — matrix operand order repaired and checked by reading, not compiled, not run on hardware
//
#include <hip/hip_runtime.h>
#include <hip/hip_bf16.h>
#include <stdint.h>

// ---------------------------------------------------------------------------
// MI455X (gfx1250) implementation.
// Heavy GEMMs use v_wmma_f32_16x16x32_f16 (f16 in, f32 accum).
// Weights pre-packed into WMMA B-operand lane layout; recurrent h/c state kept
// in LDS (f16, A-operand layout) + registers inside persistent workgroups.
// ---------------------------------------------------------------------------

typedef __attribute__((ext_vector_type(16))) _Float16 v16h;
typedef __attribute__((ext_vector_type(8)))  _Float16 v8h;
typedef __attribute__((ext_vector_type(8)))  float    v8f;

#define BS_ 32
#define T_  64
#define F_  32
#define H_  256
#define YIN_ 10

__device__ __forceinline__ float sigm(float x) { return 1.0f / (1.0f + __expf(-x)); }

// Opaque zero offset, regenerated per loop iteration. Adding it to a weight
// pointer defeats LICM (so 32+ loop-invariant weight tiles are not hoisted,
// which would blow the 128-VGPR budget and spill to scratch), while keeping
// the pointer's provenance visible to address-space inference so the backend
// emits global_load_b128 (LOADcnt only) instead of flat_load_b128
// (LOADcnt+DScnt, ties up the LDS path).
__device__ __forceinline__ size_t opaque_zero() {
    size_t z = 0;
    asm volatile("" : "+s"(z));
    return z;
}

__device__ __forceinline__ v8f wmma_f16(v16h a, v16h b, v8f c) {
    // D = A(16x32 f16) * B(32x16 f16) + C(16x16 f32)
    return __builtin_amdgcn_wmma_f32_16x16x32_f16(
        /*neg_a=*/false, a, /*neg_b=*/false, b,
        /*c_mod=*/(short)0, c, /*reuse_a=*/false, /*reuse_b=*/false);
}

// A tile (16x32, f16) from row-major f16 buffer (row=M, col=K), ldh in halves.
// Lane l holds M = m0+(l&15); halves: K = k0 + 8*(l>>4) + {0..7, 16..23}.
__device__ __forceinline__ v16h load_a_f16(const _Float16* Hbuf, int ldh, int m0, int k0, int lane) {
    const _Float16* p = Hbuf + (m0 + (lane & 15)) * ldh + k0 + 8 * (lane >> 4);
    v8h lo = *(const v8h*)(p);
    v8h hi = *(const v8h*)(p + 16);
    return __builtin_shufflevector(lo, hi, 0,1,2,3,4,5,6,7,8,9,10,11,12,13,14,15);
}

// A tile from row-major f32 global buffer, converting to f16.
__device__ __forceinline__ v16h load_a_f32(const float* A, int lda, int m0, int k0, int lane) {
    const float* p = A + (size_t)(m0 + (lane & 15)) * lda + k0 + 8 * (lane >> 4);
    v16h r;
#pragma unroll
    for (int i = 0; i < 8; i++) {
        r[i]     = (_Float16)p[i];
        r[i + 8] = (_Float16)p[16 + i];
    }
    return r;
}

// B tile (32x16, f16) from pre-packed layout [K/32][N/16][32 lanes][16 halves].
__device__ __forceinline__ v16h load_b(const _Float16* Bpk, int kt, int nt, int NT, int lane) {
    return *(const v16h*)(Bpk + ((size_t)(kt * NT + nt) * 32 + lane) * 16);
}

// ---------------------------------------------------------------------------
// Pack f32 [K][N] row-major matrix into WMMA B-operand layout (f16).
// ---------------------------------------------------------------------------
__global__ __launch_bounds__(256) void pack_kernel(const float* __restrict__ src,
                                                   _Float16* __restrict__ dst,
                                                   int K, int N) {
    int m = blockIdx.y;
    src += (size_t)m * K * N;
    dst += (size_t)m * K * N;
    int id = blockIdx.x * 256 + threadIdx.x;
    if (id >= K * N) return;
    int h    = id & 15;
    int lane = (id >> 4) & 31;
    int t2   = id >> 9;
    int NT   = N >> 4;
    int nt   = t2 % NT;
    int kt   = t2 / NT;
    int k = kt * 32 + (h & 7) + 16 * (h >> 3) + 8 * (lane >> 4);
    int n = nt * 16 + (lane & 15);
    dst[id] = (_Float16)src[(size_t)k * N + n];
}

__global__ void zero_kernel(float* p, int n) {
    int i = blockIdx.x * blockDim.x + threadIdx.x;
    if (i < n) p[i] = 0.0f;
}

// ---------------------------------------------------------------------------
// Stage 1: Y-LSTM (block 0) + 64 independent feature LSTMs (blocks 1..64).
// Persistent workgroup loops over T; h state in LDS f16, c state in VGPRs.
// 32 waves; wave = (mt, jt): mt batch-tile (0..1), jt hidden-tile (0..15).
// Each wave owns the 4 gate tiles (i,f,g,o) for its (mt,jt).
// ---------------------------------------------------------------------------
#define LDH 264  // padded row stride in halves (multiple of 8 -> 16B aligned)

__global__ __launch_bounds__(1024) void stage1_kernel(
    const float* __restrict__ Y,   const float* __restrict__ Xp,  const float* __restrict__ Xn,
    const float* __restrict__ Wy,  const float* __restrict__ by,
    const float* __restrict__ Wxp, const float* __restrict__ bxp,
    const float* __restrict__ Wxn, const float* __restrict__ bxn,
    const _Float16* __restrict__ Uy_pk, const _Float16* __restrict__ Uxp_pk,
    const _Float16* __restrict__ Uxn_pk,
    float* __restrict__ Yh, float* __restrict__ Pm, float* __restrict__ Nm) {
    __shared__ __align__(16) _Float16 Hlds[BS_ * LDH];

    int tid  = threadIdx.x;
    int lane = tid & 31;
    int wave = tid >> 5;
    int mt = wave & 1, jt = wave >> 1;
    int m0 = mt * 16;
    int g16 = lane >> 4;
    int j = jt * 16 + (lane & 15);

    int bid = blockIdx.x;
    int mode;
    int f = 0;
    const float* X = nullptr;
    const _Float16* U;
    float* Out = nullptr;
    float w4[4], b4[4], wyr[40];

    if (bid == 0) {
        mode = 0;
        U = Uy_pk;
#pragma unroll
        for (int q = 0; q < YIN_; q++)
#pragma unroll
            for (int g = 0; g < 4; g++) wyr[q * 4 + g] = Wy[q * 1024 + g * 256 + j];
#pragma unroll
        for (int g = 0; g < 4; g++) b4[g] = by[g * 256 + j];
    } else if (bid <= 32) {
        mode = 1; f = bid - 1;
        X = Xp; U = Uxp_pk + (size_t)f * 256 * 1024; Out = Pm;
#pragma unroll
        for (int g = 0; g < 4; g++) { w4[g] = Wxp[f * 1024 + g * 256 + j]; b4[g] = bxp[f * 1024 + g * 256 + j]; }
    } else {
        mode = 2; f = bid - 33;
        X = Xn; U = Uxn_pk + (size_t)f * 256 * 1024; Out = Nm;
#pragma unroll
        for (int g = 0; g < 4; g++) { w4[g] = Wxn[f * 1024 + g * 256 + j]; b4[g] = bxn[f * 1024 + g * 256 + j]; }
    }

    for (int i = tid; i < BS_ * LDH; i += 1024) Hlds[i] = (_Float16)0.0f;
    float c[8];
#pragma unroll
    for (int i = 0; i < 8; i++) c[i] = 0.0f;
    __syncthreads();

    for (int t = 0; t < T_; t++) {
        // Opaque per step: keeps weight-tile loads inside the loop (streamed
        // from L2 as global_load_b128) instead of hoisted + spilled.
        const _Float16* Ut = U + opaque_zero();

        v8f acc[4] = {};
        for (int kt = 0; kt < 8; kt++) {
            v16h bt4[4];
#pragma unroll
            for (int g = 0; g < 4; g++) bt4[g] = load_b(Ut, kt, g * 16 + jt, 64, lane);
            v16h a = load_a_f16(Hlds, LDH, m0, kt * 32, lane);
#pragma unroll
            for (int g = 0; g < 4; g++) acc[g] = wmma_f16(a, bt4[g], acc[g]);
        }
        float hv[8];
#pragma unroll
        for (int i = 0; i < 8; i++) {
            int bb = m0 + i + 8 * g16;
            float pre[4];
            if (mode == 0) {
                const float* yrow = Y + (size_t)(bb * T_ + t) * YIN_;
#pragma unroll
                for (int g = 0; g < 4; g++) {
                    float s = b4[g];
#pragma unroll
                    for (int q = 0; q < YIN_; q++) s += yrow[q] * wyr[q * 4 + g];
                    pre[g] = acc[g][i] + s;
                }
            } else {
                float xv = X[(size_t)(bb * T_ + t) * F_ + f];
#pragma unroll
                for (int g = 0; g < 4; g++) pre[g] = acc[g][i] + xv * w4[g] + b4[g];
            }
            float ig = sigm(pre[0]), fg = sigm(pre[1]);
            float gg = tanhf(pre[2]), og = sigm(pre[3]);
            c[i] = fg * c[i] + ig * gg;
            hv[i] = og * tanhf(c[i]);
        }
        __syncthreads();  // all A-operand reads of Hlds for step t done
#pragma unroll
        for (int i = 0; i < 8; i++) {
            int bb = m0 + i + 8 * g16;
            Hlds[bb * LDH + j] = (_Float16)hv[i];
            if (mode == 0)
                Yh[(size_t)(bb * T_ + t) * H_ + j] = hv[i];
            else
                atomicAdd(&Out[(size_t)(bb * T_ + t) * H_ + j], hv[i] * (1.0f / (float)F_));
        }
        __syncthreads();
    }
}

// ---------------------------------------------------------------------------
// Projection GEMMs: YG = Yh@Wm+bm [2048x1024], PG = Pm@Wpa+bpa [2048x512],
// NG = Nm@Wna+bna [2048x512]. One 16x16 tile per wave; 8 waves/block.
// ---------------------------------------------------------------------------
__global__ __launch_bounds__(256) void proj_kernel(
    const float* __restrict__ Yh, const float* __restrict__ Pm, const float* __restrict__ Nm,
    const _Float16* __restrict__ Wm_pk, const _Float16* __restrict__ Wpa_pk,
    const _Float16* __restrict__ Wna_pk,
    const float* __restrict__ bm, const float* __restrict__ bpa, const float* __restrict__ bna,
    float* __restrict__ YG, float* __restrict__ PG, float* __restrict__ NG) {
    int lane = threadIdx.x & 31;
    int wave = threadIdx.x >> 5;
    int gt = blockIdx.x * 8 + wave;

    const float* A; const _Float16* B; const float* bias; float* C; int NT;
    if (gt < 8192)       { A = Yh; B = Wm_pk;  bias = bm;  C = YG; NT = 64; }
    else if (gt < 12288) { gt -= 8192;  A = Pm; B = Wpa_pk; bias = bpa; C = PG; NT = 32; }
    else                 { gt -= 12288; A = Nm; B = Wna_pk; bias = bna; C = NG; NT = 32; }

    int mtile = gt / NT, nt = gt % NT;
    int m0 = mtile * 16;
    v8f acc = {};
    for (int kt = 0; kt < 8; kt++) {
        v16h a = load_a_f32(A, H_, m0, kt * 32, lane);
        v16h b = load_b(B, kt, nt, NT, lane);
        acc = wmma_f16(a, b, acc);
    }
    int col = nt * 16 + (lane & 15);
    int N = NT * 16;
    float bv = bias[col];
#pragma unroll
    for (int i = 0; i < 8; i++) {
        int row = m0 + i + 8 * (lane >> 4);
        C[(size_t)row * N + col] = acc[i] + bv;
    }
}

// ---------------------------------------------------------------------------
// MI-LSTM: single persistent workgroup (32 waves). Per step, wave (mt,jt)
// owns: 4 main gate tiles (Um, N=1024), 2 p-aux tiles (Upa, N=512),
// 2 n-aux tiles (Una, N=512), 1 cell-attention tile (Wa, N=256, A = c-state).
// 3-way softmax via LDS f32 atomics into sS[3][32].
// ---------------------------------------------------------------------------
__global__ __launch_bounds__(1024) void milstm_kernel(
    const float* __restrict__ YG, const float* __restrict__ PG, const float* __restrict__ NG,
    const _Float16* __restrict__ Um_pk, const _Float16* __restrict__ Upa_pk,
    const _Float16* __restrict__ Una_pk, const _Float16* __restrict__ Wa_pk,
    const float* __restrict__ ba, float* __restrict__ Hm) {
    __shared__ __align__(16) _Float16 Hlds[BS_ * LDH];
    __shared__ __align__(16) _Float16 Clds[BS_ * LDH];
    __shared__ float sS[96];

    int tid  = threadIdx.x;
    int lane = tid & 31;
    int wave = tid >> 5;
    int mt = wave & 1, jt = wave >> 1;
    int m0 = mt * 16;
    int g16 = lane >> 4;
    int j = jt * 16 + (lane & 15);
    float bav = ba[j];

    for (int i = tid; i < BS_ * LDH; i += 1024) { Hlds[i] = (_Float16)0.0f; Clds[i] = (_Float16)0.0f; }
    float c[8];
#pragma unroll
    for (int i = 0; i < 8; i++) c[i] = 0.0f;
    __syncthreads();

    for (int t = 0; t < T_; t++) {
        if (tid < 96) sS[tid] = 0.0f;
        __syncthreads();  // sS zeroed; prev-step Hlds/Clds writes visible

        size_t z = opaque_zero();
        const _Float16* Umt  = Um_pk  + z;
        const _Float16* Upat = Upa_pk + z;
        const _Float16* Unat = Una_pk + z;
        const _Float16* Wat  = Wa_pk  + z;

        v8f accm[4] = {};
        v8f accp[2] = {};
        v8f accn[2] = {};
        v8f acca = {0.f, 0.f, 0.f, 0.f, 0.f, 0.f, 0.f, 0.f};
        for (int kt = 0; kt < 8; kt++) {
            v16h aH = load_a_f16(Hlds, LDH, m0, kt * 32, lane);
            v16h aC = load_a_f16(Clds, LDH, m0, kt * 32, lane);
#pragma unroll
            for (int g = 0; g < 4; g++)
                accm[g] = wmma_f16(aH, load_b(Umt, kt, g * 16 + jt, 64, lane), accm[g]);
#pragma unroll
            for (int g = 0; g < 2; g++) {
                accp[g] = wmma_f16(aH, load_b(Upat, kt, g * 16 + jt, 32, lane), accp[g]);
                accn[g] = wmma_f16(aH, load_b(Unat, kt, g * 16 + jt, 32, lane), accn[g]);
            }
            acca = wmma_f16(aC, load_b(Wat, kt, jt, 16, lane), acca);
        }

        float f_s[8], o_s[8], l0s[8], l1s[8], l2s[8];
#pragma unroll
        for (int i = 0; i < 8; i++) {
            int bb = m0 + i + 8 * g16;
            size_t row = (size_t)(bb * T_ + t);
            float pi = accm[0][i] + YG[row * 1024 + j];
            float pf = accm[1][i] + YG[row * 1024 + 256 + j];
            float pg = accm[2][i] + YG[row * 1024 + 512 + j];
            float po = accm[3][i] + YG[row * 1024 + 768 + j];
            float ig = sigm(pi), fg = sigm(pf), og = sigm(po);
            float l0 = ig * tanhf(pg);
            float ip = sigm(accp[0][i] + PG[row * 512 + j]);
            float cp = tanhf(accp[1][i] + PG[row * 512 + 256 + j]);
            float l1 = ip * cp;
            float in_ = sigm(accn[0][i] + NG[row * 512 + j]);
            float cn  = tanhf(accn[1][i] + NG[row * 512 + 256 + j]);
            float l2 = in_ * cn;
            float att = tanhf(acca[i] + bav);
            atomicAdd(&sS[0 * 32 + bb], l0 * att);
            atomicAdd(&sS[1 * 32 + bb], l1 * att);
            atomicAdd(&sS[2 * 32 + bb], l2 * att);
            f_s[i] = fg; o_s[i] = og; l0s[i] = l0; l1s[i] = l1; l2s[i] = l2;
        }
        __syncthreads();  // scores complete; all Hlds/Clds reads done

#pragma unroll
        for (int i = 0; i < 8; i++) {
            int bb = m0 + i + 8 * g16;
            float s0 = sS[bb], s1 = sS[32 + bb], s2 = sS[64 + bb];
            float mx = fmaxf(s0, fmaxf(s1, s2));
            float e0 = __expf(s0 - mx), e1 = __expf(s1 - mx), e2 = __expf(s2 - mx);
            float inv = 1.0f / (e0 + e1 + e2);
            float lt = (e0 * l0s[i] + e1 * l1s[i] + e2 * l2s[i]) * inv;
            c[i] = f_s[i] * c[i] + lt;
            float h = o_s[i] * tanhf(c[i]);
            Hlds[bb * LDH + j] = (_Float16)h;
            Clds[bb * LDH + j] = (_Float16)c[i];
            Hm[(size_t)(bb * T_ + t) * H_ + j] = h;
        }
        __syncthreads();
    }
}

// ---------------------------------------------------------------------------
// Temporal attention scores: s[b,t] = vatt . tanh(Hm[b,t]@Watt + batt)
// ---------------------------------------------------------------------------
__global__ __launch_bounds__(256) void attscore_kernel(
    const float* __restrict__ Hm, const float* __restrict__ Watt,
    const float* __restrict__ batt, const float* __restrict__ vatt,
    float* __restrict__ Satt) {
    int bt = blockIdx.x;
    int h = threadIdx.x;
    const float* x = Hm + (size_t)bt * H_;
    float d = 0.0f;
    for (int k = 0; k < H_; k++) d += x[k] * Watt[(size_t)k * H_ + h];
    float z = tanhf(d + batt[h]) * vatt[h];
    __shared__ float red[256];
    red[h] = z;
    __syncthreads();
    for (int s = 128; s > 0; s >>= 1) {
        if (h < s) red[h] += red[h + s];
        __syncthreads();
    }
    if (h == 0) Satt[bt] = red[0];
}

// ---------------------------------------------------------------------------
// Head: softmax over t, ytilde = sum_t alpha*Hm, out = relu(ytilde@Wlin+blin)
// ---------------------------------------------------------------------------
__global__ __launch_bounds__(256) void head_kernel(
    const float* __restrict__ Hm, const float* __restrict__ Satt,
    const float* __restrict__ Wlin, const float* __restrict__ blin,
    float* __restrict__ out) {
    int b = blockIdx.x;
    int tid = threadIdx.x;
    __shared__ float alpha[T_];
    __shared__ float red[256];
    if (tid == 0) {
        float mx = -1e30f;
        for (int t = 0; t < T_; t++) mx = fmaxf(mx, Satt[b * T_ + t]);
        float s = 0.0f;
        for (int t = 0; t < T_; t++) { float e = __expf(Satt[b * T_ + t] - mx); alpha[t] = e; s += e; }
        float inv = 1.0f / s;
        for (int t = 0; t < T_; t++) alpha[t] *= inv;
    }
    __syncthreads();
    int h = tid;
    float yt = 0.0f;
    for (int t = 0; t < T_; t++) yt += alpha[t] * Hm[(size_t)(b * T_ + t) * H_ + h];
    red[h] = yt * Wlin[h];
    __syncthreads();
    for (int s = 128; s > 0; s >>= 1) {
        if (tid < s) red[tid] += red[tid + s];
        __syncthreads();
    }
    if (tid == 0) out[b] = fmaxf(red[0] + blin[0], 0.0f);
}

// ---------------------------------------------------------------------------
extern "C" void kernel_launch(void* const* d_in, const int* in_sizes, int n_in,
                              void* d_out, int out_size, void* d_ws, size_t ws_size,
                              hipStream_t stream) {
    (void)in_sizes; (void)n_in; (void)out_size; (void)ws_size;
    const float* Y    = (const float*)d_in[0];
    const float* Xp   = (const float*)d_in[1];
    const float* Xn   = (const float*)d_in[2];
    const float* Wy   = (const float*)d_in[3];
    const float* Uy   = (const float*)d_in[4];
    const float* by   = (const float*)d_in[5];
    const float* Wxp  = (const float*)d_in[6];
    const float* Uxp  = (const float*)d_in[7];
    const float* bxp  = (const float*)d_in[8];
    const float* Wxn  = (const float*)d_in[9];
    const float* Uxn  = (const float*)d_in[10];
    const float* bxn  = (const float*)d_in[11];
    const float* Wm   = (const float*)d_in[12];
    const float* Um   = (const float*)d_in[13];
    const float* bm   = (const float*)d_in[14];
    const float* Wpa  = (const float*)d_in[15];
    const float* Upa  = (const float*)d_in[16];
    const float* bpa  = (const float*)d_in[17];
    const float* Wna  = (const float*)d_in[18];
    const float* Una  = (const float*)d_in[19];
    const float* bna  = (const float*)d_in[20];
    const float* Wa   = (const float*)d_in[21];
    const float* ba   = (const float*)d_in[22];
    const float* Watt = (const float*)d_in[23];
    const float* batt = (const float*)d_in[24];
    const float* vatt = (const float*)d_in[25];
    const float* Wlin = (const float*)d_in[26];
    const float* blin = (const float*)d_in[27];
    float* out = (float*)d_out;

    // Workspace carve-up (256B aligned blocks).
    char* ws = (char*)d_ws;
    size_t off = 0;
    auto alloc = [&](size_t bytes) -> char* {
        char* p = ws + off;
        off += (bytes + 255) & ~(size_t)255;
        return p;
    };
    _Float16* Uy_pk  = (_Float16*)alloc((size_t)256 * 1024 * 2);
    _Float16* Um_pk  = (_Float16*)alloc((size_t)256 * 1024 * 2);
    _Float16* Wm_pk  = (_Float16*)alloc((size_t)256 * 1024 * 2);
    _Float16* Upa_pk = (_Float16*)alloc((size_t)256 * 512 * 2);
    _Float16* Una_pk = (_Float16*)alloc((size_t)256 * 512 * 2);
    _Float16* Wpa_pk = (_Float16*)alloc((size_t)256 * 512 * 2);
    _Float16* Wna_pk = (_Float16*)alloc((size_t)256 * 512 * 2);
    _Float16* Wa_pk  = (_Float16*)alloc((size_t)256 * 256 * 2);
    _Float16* Uxp_pk = (_Float16*)alloc((size_t)F_ * 256 * 1024 * 2);
    _Float16* Uxn_pk = (_Float16*)alloc((size_t)F_ * 256 * 1024 * 2);
    float* Yh  = (float*)alloc((size_t)BS_ * T_ * H_ * 4);
    float* Pm  = (float*)alloc((size_t)BS_ * T_ * H_ * 4);   // Pm & Nm contiguous
    float* Nm  = (float*)alloc((size_t)BS_ * T_ * H_ * 4);
    float* Hm  = (float*)alloc((size_t)BS_ * T_ * H_ * 4);
    float* YG  = (float*)alloc((size_t)BS_ * T_ * 1024 * 4);
    float* PG  = (float*)alloc((size_t)BS_ * T_ * 512 * 4);
    float* NG  = (float*)alloc((size_t)BS_ * T_ * 512 * 4);
    float* Satt = (float*)alloc((size_t)BS_ * T_ * 4);

    // 1) Pack weights to WMMA f16 B layout.
    pack_kernel<<<dim3(1024, 1), 256, 0, stream>>>(Uy,  Uy_pk,  256, 1024);
    pack_kernel<<<dim3(1024, 1), 256, 0, stream>>>(Um,  Um_pk,  256, 1024);
    pack_kernel<<<dim3(1024, 1), 256, 0, stream>>>(Wm,  Wm_pk,  256, 1024);
    pack_kernel<<<dim3(512, 1),  256, 0, stream>>>(Upa, Upa_pk, 256, 512);
    pack_kernel<<<dim3(512, 1),  256, 0, stream>>>(Una, Una_pk, 256, 512);
    pack_kernel<<<dim3(512, 1),  256, 0, stream>>>(Wpa, Wpa_pk, 256, 512);
    pack_kernel<<<dim3(512, 1),  256, 0, stream>>>(Wna, Wna_pk, 256, 512);
    pack_kernel<<<dim3(256, 1),  256, 0, stream>>>(Wa,  Wa_pk,  256, 256);
    pack_kernel<<<dim3(1024, F_), 256, 0, stream>>>(Uxp, Uxp_pk, 256, 1024);
    pack_kernel<<<dim3(1024, F_), 256, 0, stream>>>(Uxn, Uxn_pk, 256, 1024);

    // 2) Zero the feature-mean accumulators (Pm,Nm contiguous).
    zero_kernel<<<4096, 256, 0, stream>>>(Pm, 2 * BS_ * T_ * H_);

    // 3) Recurrent stage 1: Y-LSTM + 64 feature LSTMs (persistent WGs).
    stage1_kernel<<<65, 1024, 0, stream>>>(Y, Xp, Xn, Wy, by, Wxp, bxp, Wxn, bxn,
                                           Uy_pk, Uxp_pk, Uxn_pk, Yh, Pm, Nm);

    // 4) Input projections for the MI-LSTM.
    proj_kernel<<<2048, 256, 0, stream>>>(Yh, Pm, Nm, Wm_pk, Wpa_pk, Wna_pk,
                                          bm, bpa, bna, YG, PG, NG);

    // 5) MI-LSTM recurrence (persistent WG).
    milstm_kernel<<<1, 1024, 0, stream>>>(YG, PG, NG, Um_pk, Upa_pk, Una_pk, Wa_pk, ba, Hm);

    // 6) Temporal attention + output head.
    attscore_kernel<<<BS_ * T_, 256, 0, stream>>>(Hm, Watt, batt, vatt, Satt);
    head_kernel<<<BS_, 256, 0, stream>>>(Hm, Satt, Wlin, blin, out);
}
